// Cross_Graph_32272384262430
// MI455X (gfx1250) — compile-verified, matrix operand order checked by reading, not run on hardware
//
#include <hip/hip_runtime.h>

// ---------------------------------------------------------------------------
// CDNA5 (gfx1250, wave32). bf16 WMMA 16x16x32 for all GEMMs (f32 accumulate),
// TDM (tensor_load_to_lds) staging for flash-attention K/V tiles, fused FF
// holding the 16x2048 hidden tile in LDS.
// ---------------------------------------------------------------------------

typedef __attribute__((ext_vector_type(16))) __bf16 v16bf;
typedef __attribute__((ext_vector_type(8)))  float  v8f;
typedef __attribute__((ext_vector_type(4)))  unsigned int u32x4;
typedef __attribute__((ext_vector_type(8)))  int  i32x8;
typedef __attribute__((ext_vector_type(4)))  int  i32x4;

#define HID    64
#define DKDIM  32
#define FFDIM  2048
#define DNODES 1024
#define GSEG   4
#define NROWS  16384            // B * DNODES * GSEG

__device__ __forceinline__ v8f wmma_bf16(v16bf a, v16bf b, v8f c) {
  return __builtin_amdgcn_wmma_f32_16x16x32_bf16(false, a, false, b, (short)0, c,
                                                 false, false);
}

// ---------------------------------------------------------------------------
// TDM: 1-D contiguous copy global -> LDS. D# built per ISA 08_async_tensor
// 8.3/8.4: group0 {count=1, lds_addr, global_addr, type=2}; group1 sets
// data_size=4B, tensor_dim0 = tile_dim0 = ndwords, tensor_dim1 = tile_dim1 = 1.
// This toolchain exposes the 6-arg builtin (clang-23 form): groups 2/3 and the
// trailing i32x8 are zero for a <=2-D tensor.
// ---------------------------------------------------------------------------
__device__ __forceinline__ void tdm_load_1d(unsigned lds_off, const void* gptr,
                                            unsigned ndwords) {
  unsigned long long ga = (unsigned long long)(uintptr_t)gptr;
  u32x4 g0;
  g0[0] = 1u;                                            // count=1
  g0[1] = lds_off;                                       // lds_addr (bytes)
  g0[2] = (unsigned)(ga & 0xffffffffull);                // global_addr[31:0]
  g0[3] = (unsigned)((ga >> 32) & 0x1ffffffull) | (2u << 30); // addr[56:32]|type=2
  i32x8 g1;
  g1[0] = (2 << 16);                                     // data_size=2 (4B)
  g1[1] = (int)((ndwords & 0xffffu) << 16);              // tensor_dim0[15:0]
  g1[2] = (int)(((ndwords >> 16) & 0xffffu) | (1u << 16)); // td0[31:16], td1=1
  g1[3] = (int)((ndwords & 0xffffu) << 16);              // tile_dim0
  g1[4] = 1;                                             // tile_dim1=1, tile_dim2=0
  g1[5] = (int)ndwords;                                  // tensor_dim0_stride lo
  g1[6] = 0;
  g1[7] = 0;
  i32x4 zero4 = {0, 0, 0, 0};
  i32x8 zero8 = {0, 0, 0, 0, 0, 0, 0, 0};
  __builtin_amdgcn_tensor_load_to_lds(g0, g1, zero4, zero4, zero8, 0);
}

// Fragment loader (A-matrix or N-major B-matrix): lane l holds "row" (l&15),
// K split by lane-half per ISA 7.12.2: elems[0:8)=K kh..kh+7, [8:16)=K 16+kh..
__device__ __forceinline__ v16bf frag_f32(const float* p0, int ld, int lane, int kb) {
  const float* p = p0 + (size_t)(lane & 15) * ld + kb;
  int kh = (lane >> 4) * 8;
  v16bf r;
#pragma unroll
  for (int i = 0; i < 8; ++i) r[i] = (__bf16)p[kh + i];
#pragma unroll
  for (int i = 0; i < 8; ++i) r[8 + i] = (__bf16)p[16 + kh + i];
  return r;
}
__device__ __forceinline__ v16bf frag_bf16(const __bf16* p0, int ld, int lane, int kb) {
  const __bf16* p = p0 + (size_t)(lane & 15) * ld + kb;
  int kh = (lane >> 4) * 8;
  v16bf r;
#pragma unroll
  for (int i = 0; i < 8; ++i) r[i] = p[kh + i];
#pragma unroll
  for (int i = 0; i < 8; ++i) r[8 + i] = p[16 + kh + i];
  return r;
}
// B-fragment from K-major storage [k][n] (leading dim ldn): col = n0 + (lane&15)
__device__ __forceinline__ v16bf frag_bf16_kmajor(const __bf16* p0, int ldn, int lane, int n0) {
  int n = n0 + (lane & 15);
  int kh = (lane >> 4) * 8;
  v16bf r;
#pragma unroll
  for (int i = 0; i < 8; ++i) r[i] = p0[(kh + i) * ldn + n];
#pragma unroll
  for (int i = 0; i < 8; ++i) r[8 + i] = p0[(16 + kh + i) * ldn + n];
  return r;
}

// ---------------------------------------------------------------------------
// Generic WMMA GEMM: C[M,N] = A[M,K] @ W[N,K]^T (+ bias). One wave = one 16x16
// tile. Compile-time variants: SWIZ = output rows remapped [b,d,g]->[b,g,d];
// OUTBF = bf16 output (no bias) vs f32 output (+bias).
// ---------------------------------------------------------------------------
template <bool SWIZ, bool OUTBF>
__global__ __launch_bounds__(32) void wmma_gemm(
    const float* __restrict__ A, const float* __restrict__ W,
    const float* __restrict__ bias, float* __restrict__ Cf,
    __bf16* __restrict__ Cb, int N, int K) {
  int lane = threadIdx.x;
  int m0 = blockIdx.x * 16;
  int n0 = blockIdx.y * 16;
  v8f acc = {};
  for (int kb = 0; kb < K; kb += 32) {
    v16bf a = frag_f32(A + (size_t)m0 * K, K, lane, kb);
    v16bf b = frag_f32(W + (size_t)n0 * K, K, lane, kb);
    acc = wmma_bf16(a, b, acc);
  }
  int row0 = (lane >> 4) * 8, col = lane & 15;
  float bv = OUTBF ? 0.f : bias[n0 + col];
#pragma unroll
  for (int j = 0; j < 8; ++j) {
    int r = m0 + row0 + j;
    int ro = r;
    if (SWIZ) {
      int bb = r >> 12;                 // / (DNODES*GSEG)
      int rem = r & (DNODES * GSEG - 1);
      int d = rem >> 2;                 // / GSEG
      int gg = rem & (GSEG - 1);
      ro = (bb * GSEG + gg) * DNODES + d;
    }
    float v = acc[j] + bv;
    if (OUTBF) Cb[(size_t)ro * N + n0 + col] = (__bf16)v;
    else       Cf[(size_t)ro * N + n0 + col] = v;
  }
}

// ---------------------------------------------------------------------------
// Fused FF: Y = relu(X @ W1^T + b1) @ W2^T + b2, 16 rows per block, hidden
// tile (16 x 2048 bf16 = 64 KB) staged in LDS. 4 waves / block.
// ---------------------------------------------------------------------------
__global__ __launch_bounds__(128) void ff_fused(
    const float* __restrict__ X, const float* __restrict__ W1,
    const float* __restrict__ B1, const float* __restrict__ W2,
    const float* __restrict__ B2, float* __restrict__ Y) {
  __shared__ __align__(16) __bf16 hs[16 * FFDIM];
  int wave = threadIdx.x >> 5;
  int lane = threadIdx.x & 31;
  int m0 = blockIdx.x * 16;
  int row0 = (lane >> 4) * 8, col = lane & 15;
  v16bf xa0 = frag_f32(X + (size_t)m0 * HID, HID, lane, 0);
  v16bf xa1 = frag_f32(X + (size_t)m0 * HID, HID, lane, 32);
  for (int nt = wave; nt < FFDIM / 16; nt += 4) {
    v8f c = {};
    c = wmma_bf16(xa0, frag_f32(W1 + (size_t)nt * 16 * HID, HID, lane, 0), c);
    c = wmma_bf16(xa1, frag_f32(W1 + (size_t)nt * 16 * HID, HID, lane, 32), c);
    float bb = B1[nt * 16 + col];
#pragma unroll
    for (int j = 0; j < 8; ++j) {
      float v = c[j] + bb;
      hs[(row0 + j) * FFDIM + nt * 16 + col] = (__bf16)(v > 0.f ? v : 0.f);
    }
  }
  __syncthreads();
  v8f acc = {};
  const float* w2 = W2 + (size_t)wave * 16 * FFDIM;
  for (int kc = 0; kc < FFDIM; kc += 32)
    acc = wmma_bf16(frag_bf16(hs, FFDIM, lane, kc), frag_f32(w2, FFDIM, lane, kc), acc);
  float bb2 = B2[wave * 16 + col];
#pragma unroll
  for (int j = 0; j < 8; ++j)
    Y[(size_t)(m0 + row0 + j) * HID + wave * 16 + col] = acc[j] + bb2;
}

// ---------------------------------------------------------------------------
// Flash attention: per (batch, 16-query tile). dk=32 (one WMMA K), dv=64.
// K/V chunks DMA'd into LDS by the Tensor Data Mover (TENSORcnt-tracked);
// online softmax via 16-lane shfl reductions. Single wave per workgroup:
// same-wave LDS ops are in-order, so no barriers needed.
// ---------------------------------------------------------------------------
__global__ __launch_bounds__(32) void flash_attn(
    const __bf16* __restrict__ Q, const __bf16* __restrict__ Kd,
    const __bf16* __restrict__ Vd, float* __restrict__ O, int nkv, float scale) {
  int batch = blockIdx.y;
  int q0 = blockIdx.x * 16;
  int lane = threadIdx.x;
  __shared__ __align__(16) __bf16 ks[32 * DKDIM];
  __shared__ __align__(16) __bf16 vs[32 * HID];
  __shared__ __align__(16) __bf16 ps[16 * 32];
  unsigned ks_off = (unsigned)(uintptr_t)(&ks[0]);   // flat addr low 32b = LDS byte offset
  unsigned vs_off = (unsigned)(uintptr_t)(&vs[0]);
  v16bf qa = frag_bf16(Q + ((size_t)batch * nkv + q0) * DKDIM, DKDIM, lane, 0);
  v8f a0 = {}, a1 = {}, a2 = {}, a3 = {};
  float m[8], l[8];
#pragma unroll
  for (int j = 0; j < 8; ++j) { m[j] = -1e30f; l[j] = 0.f; }
  int row0 = (lane >> 4) * 8, col = lane & 15;
  for (int kb = 0; kb < nkv; kb += 32) {
    // WAR guard: previous iteration's DS reads must drain before DMA overwrite
    asm volatile("s_wait_dscnt 0x0" ::: "memory");
    tdm_load_1d(ks_off, Kd + ((size_t)batch * nkv + kb) * DKDIM, (32 * DKDIM * 2) / 4);
    tdm_load_1d(vs_off, Vd + ((size_t)batch * nkv + kb) * HID, (32 * HID * 2) / 4);
    __builtin_amdgcn_s_wait_tensorcnt(0);
    v8f s0 = {}, s1 = {};
    s0 = wmma_bf16(qa, frag_bf16(ks, DKDIM, lane, 0), s0);
    s1 = wmma_bf16(qa, frag_bf16(ks + 16 * DKDIM, DKDIM, lane, 0), s1);
#pragma unroll
    for (int j = 0; j < 8; ++j) {
      float e0 = s0[j] * scale, e1 = s1[j] * scale;
      float mx = fmaxf(e0, e1);
#pragma unroll
      for (int sh = 1; sh < 16; sh <<= 1) mx = fmaxf(mx, __shfl_xor(mx, sh, 32));
      float mn = fmaxf(m[j], mx);
      float corr = __expf(m[j] - mn);
      float p0 = __expf(e0 - mn), p1 = __expf(e1 - mn);
      float ss = p0 + p1;
#pragma unroll
      for (int sh = 1; sh < 16; sh <<= 1) ss += __shfl_xor(ss, sh, 32);
      l[j] = l[j] * corr + ss;
      m[j] = mn;
      a0[j] *= corr; a1[j] *= corr; a2[j] *= corr; a3[j] *= corr;
      ps[(row0 + j) * 32 + col] = (__bf16)p0;
      ps[(row0 + j) * 32 + 16 + col] = (__bf16)p1;
    }
    v16bf pa = frag_bf16(ps, 32, lane, 0);
    a0 = wmma_bf16(pa, frag_bf16_kmajor(vs, HID, lane, 0), a0);
    a1 = wmma_bf16(pa, frag_bf16_kmajor(vs, HID, lane, 16), a1);
    a2 = wmma_bf16(pa, frag_bf16_kmajor(vs, HID, lane, 32), a2);
    a3 = wmma_bf16(pa, frag_bf16_kmajor(vs, HID, lane, 48), a3);
  }
#pragma unroll
  for (int j = 0; j < 8; ++j) {
    float invl = 1.0f / l[j];
    size_t ro = ((size_t)batch * nkv + q0 + row0 + j) * HID + col;
    O[ro]      = a0[j] * invl;
    O[ro + 16] = a1[j] * invl;
    O[ro + 32] = a2[j] * invl;
    O[ro + 48] = a3[j] * invl;
  }
}

// ---------------------------------------------------------------------------
// Patch embedding + sinusoidal PE.  row = ((b*D + d)*G + g), thread = h.
// ---------------------------------------------------------------------------
__global__ __launch_bounds__(64) void patch_embed(
    const float* __restrict__ x, const float* __restrict__ pw,
    const float* __restrict__ pb, float* __restrict__ z) {
  int row = blockIdx.x;
  int h = threadIdx.x;
  int g = row & 3;
  int bd = row >> 2;
  int d = bd & (DNODES - 1);
  int b = bd >> 10;
  float acc = pb[h];
#pragma unroll
  for (int l = 0; l < 3; ++l)
    acc += x[(size_t)(b * 12 + g * 3 + l) * DNODES + d] * pw[h * 3 + l];
  int i2 = (h >> 1) * 2;
  float div = __expf(-(float)i2 * (9.2103403720f / 64.f));
  float ang = (float)g * div;
  acc += (h & 1) ? __cosf(ang) : __sinf(ang);
  z[(size_t)row * HID + h] = acc;
}

// ---------------------------------------------------------------------------
// Temporal micro-attention: seq=4, 8 heads of dim 8. One thread per (bd,head).
// ---------------------------------------------------------------------------
__global__ __launch_bounds__(256) void micro_attn(
    const float* __restrict__ QKV, float* __restrict__ O) {
  int t = blockIdx.x * blockDim.x + threadIdx.x;
  if (t >= 4096 * 8) return;
  int h = t & 7;
  int bd = t >> 3;
  const float* base = QKV + (size_t)bd * 4 * 192 + h * 8;
  float q[4][8], k[4][8], v[4][8];
#pragma unroll
  for (int g = 0; g < 4; ++g)
#pragma unroll
    for (int e = 0; e < 8; ++e) {
      q[g][e] = base[g * 192 + e];
      k[g][e] = base[g * 192 + 64 + e];
      v[g][e] = base[g * 192 + 128 + e];
    }
  const float scale = 0.3535533906f;   // 1/sqrt(8)
#pragma unroll
  for (int gq = 0; gq < 4; ++gq) {
    float s[4], mx = -1e30f;
#pragma unroll
    for (int gk = 0; gk < 4; ++gk) {
      float dd = 0.f;
#pragma unroll
      for (int e = 0; e < 8; ++e) dd += q[gq][e] * k[gk][e];
      s[gk] = dd * scale;
      mx = fmaxf(mx, s[gk]);
    }
    float sum = 0.f;
#pragma unroll
    for (int gk = 0; gk < 4; ++gk) { s[gk] = __expf(s[gk] - mx); sum += s[gk]; }
    float inv = 1.0f / sum;
#pragma unroll
    for (int e = 0; e < 8; ++e) {
      float o = 0.f;
#pragma unroll
      for (int gk = 0; gk < 4; ++gk) o += s[gk] * v[gk][e];
      O[((size_t)bd * 4 + gq) * HID + h * 8 + e] = o * inv;
    }
  }
}

// ---------------------------------------------------------------------------
// LayerNorm over 64 with residual. MODE 0: Y = LN(X + R); MODE 1: Y = LN(X)+R.
// SWIZ: read X at row [b,g,d] while Y/R rows are [b,d,g]. One wave per row.
// ---------------------------------------------------------------------------
template <int MODE, bool SWIZ>
__global__ __launch_bounds__(256) void ln_res(
    const float* __restrict__ X, const float* __restrict__ R,
    const float* __restrict__ gma, const float* __restrict__ bta,
    float* __restrict__ Y, int rows) {
  int r = blockIdx.x * (blockDim.x >> 5) + (threadIdx.x >> 5);
  if (r >= rows) return;
  int lane = threadIdx.x & 31;
  int xr = r;
  if (SWIZ) {
    int bb = r >> 12;
    int rem = r & (DNODES * GSEG - 1);
    int d = rem >> 2;
    int gg = rem & (GSEG - 1);
    xr = (bb * GSEG + gg) * DNODES + d;
  }
  const float* xp = X + (size_t)xr * HID;
  const float* rp = R + (size_t)r * HID;
  float v0 = xp[lane], v1 = xp[lane + 32];
  float r0 = rp[lane], r1 = rp[lane + 32];
  if (MODE == 0) { v0 += r0; v1 += r1; }
  float s = v0 + v1;
#pragma unroll
  for (int m = 1; m < 32; m <<= 1) s += __shfl_xor(s, m, 32);
  float mean = s * (1.0f / HID);
  float d0 = v0 - mean, d1 = v1 - mean;
  float vs = d0 * d0 + d1 * d1;
#pragma unroll
  for (int m = 1; m < 32; m <<= 1) vs += __shfl_xor(vs, m, 32);
  float inv = rsqrtf(vs * (1.0f / HID) + 1e-5f);
  float y0 = d0 * inv * gma[lane] + bta[lane];
  float y1 = d1 * inv * gma[lane + 32] + bta[lane + 32];
  if (MODE == 1) { y0 += r0; y1 += r1; }
  Y[(size_t)r * HID + lane] = y0;
  Y[(size_t)r * HID + lane + 32] = y1;
}

// Mean-pool over the 4 patches and concat the two streams -> enc [4,1024,128]
__global__ __launch_bounds__(256) void pool_concat(
    const float* __restrict__ z0, const float* __restrict__ z1,
    float* __restrict__ enc) {
  int idx = blockIdx.x * blockDim.x + threadIdx.x;
  if (idx >= 4 * DNODES * HID) return;
  int h = idx & (HID - 1);
  int bd = idx >> 6;
  const float* p0 = z0 + (size_t)bd * (GSEG * HID) + h;
  const float* p1 = z1 + (size_t)bd * (GSEG * HID) + h;
  float s0 = 0.f, s1 = 0.f;
#pragma unroll
  for (int g = 0; g < GSEG; ++g) { s0 += p0[g * HID]; s1 += p1[g * HID]; }
  enc[(size_t)bd * 128 + h] = s0 * 0.25f;
  enc[(size_t)bd * 128 + HID + h] = s1 * 0.25f;
}

// ---------------------------------------------------------------------------
extern "C" void kernel_launch(void* const* d_in, const int* in_sizes, int n_in,
                              void* d_out, int out_size, void* d_ws, size_t ws_size,
                              hipStream_t stream) {
  const float* traffic = (const float*)d_in[0];
  const float* user    = (const float*)d_in[1];
  const float* patch_w = (const float*)d_in[2];
  const float* patch_b = (const float*)d_in[3];
  const float* mha_w   = (const float*)d_in[4];
  const float* mha_b   = (const float*)d_in[5];
  const float* out_w   = (const float*)d_in[6];
  const float* out_b   = (const float*)d_in[7];
  const float* ff1_w   = (const float*)d_in[8];
  const float* ff1_b   = (const float*)d_in[9];
  const float* ff2_w   = (const float*)d_in[10];
  const float* ff2_b   = (const float*)d_in[11];
  const float* ln1_g   = (const float*)d_in[12];
  const float* ln1_b   = (const float*)d_in[13];
  const float* ln2_g   = (const float*)d_in[14];
  const float* ln2_b   = (const float*)d_in[15];
  const float* ga_wq   = (const float*)d_in[16];
  const float* ga_wk   = (const float*)d_in[17];
  const float* ga_wv   = (const float*)d_in[18];
  const float* ga_f1w  = (const float*)d_in[19];
  const float* ga_f1b  = (const float*)d_in[20];
  const float* ga_f2w  = (const float*)d_in[21];
  const float* ga_f2b  = (const float*)d_in[22];
  const float* ga_l1g  = (const float*)d_in[23];
  const float* ga_l1b  = (const float*)d_in[24];
  const float* ga_l2g  = (const float*)d_in[25];
  const float* ga_l2b  = (const float*)d_in[26];
  const float* ca_wq   = (const float*)d_in[27];
  const float* ca_wk   = (const float*)d_in[28];
  const float* ca_wv   = (const float*)d_in[29];

  // workspace layout (floats, then bf16 region)
  float* wsf = (float*)d_ws;
  float* z0  = wsf;                       // 1,048,576
  float* z1  = z0  + 1048576;             // 1,048,576
  float* qkv = z1  + 1048576;             // 3,145,728
  float* t1  = qkv + 3145728;             // 1,048,576
  float* t2  = t1  + 1048576;             // 1,048,576
  float* att = t2  + 1048576;             // 1,048,576
  float* enc = att + 1048576;             //   524,288
  __bf16* qb = (__bf16*)(enc + 524288);   // 524,288 bf16
  __bf16* kb = qb + 524288;               // 524,288
  __bf16* vb = kb + 524288;               // 1,048,576
  __bf16* cq = vb + 1048576;              // 131,072
  __bf16* ck = cq + 131072;               // 131,072
  __bf16* cv = ck + 131072;               // 262,144

  const float inv_sqrt_dk = 0.1767766953f;   // 1/sqrt(32)

  for (int s = 0; s < 2; ++s) {
    float* z = s ? z1 : z0;
    patch_embed<<<dim3(NROWS), dim3(64), 0, stream>>>(
        s ? user : traffic, patch_w + s * 192, patch_b + s * 64, z);
    for (int i = 0; i < 2; ++i) {
      int sl = s * 2 + i;
      // ---- temporal encoder layer ----
      wmma_gemm<false, false><<<dim3(1024, 12), 32, 0, stream>>>(
          z, mha_w + (size_t)sl * 192 * 64, mha_b + sl * 192, qkv, nullptr, 192, 64);
      micro_attn<<<dim3(128), 256, 0, stream>>>(qkv, t1);
      wmma_gemm<false, false><<<dim3(1024, 4), 32, 0, stream>>>(
          t1, out_w + (size_t)sl * 64 * 64, out_b + sl * 64, t2, nullptr, 64, 64);
      ln_res<0, false><<<dim3(2048), 256, 0, stream>>>(
          t2, z, ln1_g + sl * 64, ln1_b + sl * 64, z, NROWS);
      ff_fused<<<dim3(1024), 128, 0, stream>>>(
          z, ff1_w + (size_t)sl * 2048 * 64, ff1_b + sl * 2048,
          ff2_w + (size_t)sl * 64 * 2048, ff2_b + sl * 64, t1);
      ln_res<0, false><<<dim3(2048), 256, 0, stream>>>(
          t1, z, ln2_g + sl * 64, ln2_b + sl * 64, z, NROWS);
      // ---- spatial graph attention ----
      wmma_gemm<true, true><<<dim3(1024, 2), 32, 0, stream>>>(
          z, ga_wq + (size_t)sl * 32 * 64, nullptr, nullptr, qb, 32, 64);
      wmma_gemm<true, true><<<dim3(1024, 2), 32, 0, stream>>>(
          z, ga_wk + (size_t)sl * 32 * 64, nullptr, nullptr, kb, 32, 64);
      wmma_gemm<true, true><<<dim3(1024, 4), 32, 0, stream>>>(
          z, ga_wv + (size_t)sl * 64 * 64, nullptr, nullptr, vb, 64, 64);
      flash_attn<<<dim3(64, 16), 32, 0, stream>>>(qb, kb, vb, att, DNODES, inv_sqrt_dk);
      ln_res<1, true><<<dim3(2048), 256, 0, stream>>>(
          att, z, ga_l1g + sl * 64, ga_l1b + sl * 64, t2, NROWS);
      ff_fused<<<dim3(1024), 128, 0, stream>>>(
          t2, ga_f1w + (size_t)sl * 2048 * 64, ga_f1b + sl * 2048,
          ga_f2w + (size_t)sl * 64 * 2048, ga_f2b + sl * 64, t1);
      ln_res<0, false><<<dim3(2048), 256, 0, stream>>>(
          t1, t2, ga_l2g + sl * 64, ga_l2b + sl * 64, z, NROWS);
    }
  }

  // ---- pooling + final cross attention ----
  pool_concat<<<dim3(1024), 256, 0, stream>>>(z0, z1, enc);
  wmma_gemm<false, true><<<dim3(256, 2), 32, 0, stream>>>(
      enc, ca_wq, nullptr, nullptr, cq, 32, 128);
  wmma_gemm<false, true><<<dim3(256, 2), 32, 0, stream>>>(
      enc, ca_wk, nullptr, nullptr, ck, 32, 128);
  wmma_gemm<false, true><<<dim3(256, 4), 32, 0, stream>>>(
      enc, ca_wv, nullptr, nullptr, cv, 64, 128);
  flash_attn<<<dim3(64, 4), 32, 0, stream>>>(cq, ck, cv, (float*)d_out, DNODES, inv_sqrt_dk);
}
